// NeuronGAT_10101763080487
// MI455X (gfx1250) — matrix operand order repaired and checked by reading, not compiled, not run on hardware
//
#include <hip/hip_runtime.h>
#include <hip/hip_bf16.h>

#define NEG_SLOPE 0.2f

typedef __attribute__((ext_vector_type(16))) _Float16 v16h;
typedef __attribute__((ext_vector_type(8)))  _Float16 v8h;
typedef __attribute__((ext_vector_type(8)))  float    v8f;

__device__ __forceinline__ float lrelu(float v) { return v > 0.f ? v : NEG_SLOPE * v; }

// Order-preserving float<->uint encoding for atomicMax on floats.
__device__ __forceinline__ unsigned fenc(float f) {
    unsigned u = __float_as_uint(f);
    return (u & 0x80000000u) ? ~u : (u | 0x80000000u);
}
__device__ __forceinline__ float fdec(unsigned u) {
    u = (u & 0x80000000u) ? (u & 0x7FFFFFFFu) : ~u;
    return __uint_as_float(u);
}

// ---------------- Cross-lane reductions via DPP (no LDS traffic) ----------------
// xor-1/2/4 inside groups of 8 via DPP8; xor-8 inside a 16-lane row via DPP16 row_ror:8.
template <int M>
__device__ __forceinline__ float dpp8_xorf(float x) {
    constexpr unsigned sel = (unsigned)((0 ^ M) | ((1 ^ M) << 3) | ((2 ^ M) << 6) |
                                        ((3 ^ M) << 9) | ((4 ^ M) << 12) | ((5 ^ M) << 15) |
                                        ((6 ^ M) << 18) | ((7 ^ M) << 21));
    return __int_as_float(__builtin_amdgcn_mov_dpp8(__float_as_int(x), sel));
}
__device__ __forceinline__ float dpp_ror8f(float x) {
    int xi = __float_as_int(x);
    return __int_as_float(__builtin_amdgcn_update_dpp(xi, xi, 0x128, 0xf, 0xf, false));
}
// Reduce across the 16 lanes of each half-wave (rows live per half).
__device__ __forceinline__ float row16_max(float x) {
    x = fmaxf(x, dpp8_xorf<1>(x));
    x = fmaxf(x, dpp8_xorf<2>(x));
    x = fmaxf(x, dpp8_xorf<4>(x));
    x = fmaxf(x, dpp_ror8f(x));
    return x;
}
__device__ __forceinline__ float row16_sum(float x) {
    x += dpp8_xorf<1>(x);
    x += dpp8_xorf<2>(x);
    x += dpp8_xorf<4>(x);
    x += dpp_ror8f(x);
    return x;
}

// ---------------- WMMA tile loaders (wave32, v_wmma_f32_16x16x32_f16) ----------------
// A operand 16(M)x32(K): lane&15 = M row; element j -> K = j + 16*(lane>>4).
// One run of 16 consecutive floats per lane -> 4x float4 loads. Caller guarantees full tiles.
__device__ __forceinline__ v16h load_a_f4(const float* __restrict__ src, int ld, int m0, int k0,
                                          float scale) {
    const int lane = threadIdx.x & 31, r = lane & 15, hi = lane >> 4;
    const float* base = src + (size_t)(m0 + r) * ld + k0 + 16 * hi;
    float4 a0 = *(const float4*)(base + 0);
    float4 a1 = *(const float4*)(base + 4);
    float4 a2 = *(const float4*)(base + 8);
    float4 a3 = *(const float4*)(base + 12);
    v16h a;
    a[0]  = (_Float16)(a0.x * scale); a[1]  = (_Float16)(a0.y * scale);
    a[2]  = (_Float16)(a0.z * scale); a[3]  = (_Float16)(a0.w * scale);
    a[4]  = (_Float16)(a1.x * scale); a[5]  = (_Float16)(a1.y * scale);
    a[6]  = (_Float16)(a1.z * scale); a[7]  = (_Float16)(a1.w * scale);
    a[8]  = (_Float16)(a2.x * scale); a[9]  = (_Float16)(a2.y * scale);
    a[10] = (_Float16)(a2.z * scale); a[11] = (_Float16)(a2.w * scale);
    a[12] = (_Float16)(a3.x * scale); a[13] = (_Float16)(a3.y * scale);
    a[14] = (_Float16)(a3.z * scale); a[15] = (_Float16)(a3.w * scale);
    return a;
}
// B operand 32(K)x16(N) from row-major [K][N] weights: lane&15 = N col; element j -> K = j + 16*hi.
__device__ __forceinline__ v16h load_b_km(const float* __restrict__ src, int ld, int k0, int n0) {
    const int lane = threadIdx.x & 31, r = lane & 15, hi = lane >> 4;
    const float* base = src + (size_t)(k0 + 16 * hi) * ld + n0 + r;
    v16h b;
#pragma unroll
    for (int j = 0; j < 16; ++j) b[j] = (_Float16)base[(size_t)j * ld];
    return b;
}
// B operand built from 16 consecutive floats per lane (for S = Q @ K^T).
__device__ __forceinline__ v16h load_b_contig(const float* __restrict__ base) {
    float4 c0 = *(const float4*)(base + 0), c1 = *(const float4*)(base + 4);
    float4 c2 = *(const float4*)(base + 8), c3 = *(const float4*)(base + 12);
    v16h b;
    b[0]=(_Float16)c0.x; b[1]=(_Float16)c0.y; b[2]=(_Float16)c0.z; b[3]=(_Float16)c0.w;
    b[4]=(_Float16)c1.x; b[5]=(_Float16)c1.y; b[6]=(_Float16)c1.z; b[7]=(_Float16)c1.w;
    b[8]=(_Float16)c2.x; b[9]=(_Float16)c2.y; b[10]=(_Float16)c2.z; b[11]=(_Float16)c2.w;
    b[12]=(_Float16)c3.x; b[13]=(_Float16)c3.y; b[14]=(_Float16)c3.z; b[15]=(_Float16)c3.w;
    return b;
}

// ---------------- GEMM: C[M,N] = A[M,K] @ B[K,N] (+bias)(+leaky); NT 16-col tiles per wave ----
template <int NT>
__global__ __launch_bounds__(32) void gemm_wmma_k(const float* __restrict__ A,
                                                  const float* __restrict__ B,
                                                  const float* __restrict__ bias,
                                                  float* __restrict__ C,
                                                  int M, int K, int N, int act) {
    const int m0 = blockIdx.x * 16;
    const int n0 = blockIdx.y * (16 * NT);
    const int lane = threadIdx.x & 31, r = lane & 15, hi = lane >> 4;
    v8f acc[NT];
#pragma unroll
    for (int t = 0; t < NT; ++t) acc[t] = (v8f){};
    for (int k0 = 0; k0 < K; k0 += 32) {
        // speculative prefetch of next A tile (global_prefetch_b8; dropped if OOB)
        __builtin_prefetch(A + (size_t)(m0 + r) * K + k0 + 32 + 8 * hi, 0, 3);
        v16h a = load_a_f4(A, K, m0, k0, 1.f);
#pragma unroll
        for (int t = 0; t < NT; ++t) {
            v16h b = load_b_km(B, N, k0, n0 + 16 * t);
            acc[t] = __builtin_amdgcn_wmma_f32_16x16x32_f16(false, a, false, b, (short)0,
                                                            acc[t], false, false);
        }
    }
#pragma unroll
    for (int t = 0; t < NT; ++t) {
        int col = n0 + 16 * t + r;
        float bv = bias ? bias[col] : 0.f;
#pragma unroll
        for (int i = 0; i < 8; ++i) {
            float v = acc[t][i] + bv;
            if (act) v = lrelu(v);
            C[(size_t)(m0 + i + 8 * hi) * N + col] = v;
        }
    }
}

// ---------------- GAT attention coefficients ----------------
__global__ void gat_alpha_k(const float* __restrict__ hw, const float* __restrict__ as,
                            const float* __restrict__ ad, float* __restrict__ asrc,
                            float* __restrict__ adst, int N, int heads) {
    int idx = blockIdx.x * blockDim.x + threadIdx.x;
    if (idx >= N * heads) return;
    int n = idx / heads, h = idx % heads;
    const float* row = hw + (size_t)n * heads * 64 + h * 64;
    float s = 0.f, d = 0.f;
#pragma unroll 8
    for (int c = 0; c < 64; ++c) {
        float v = row[c];
        s += v * as[h * 64 + c];
        d += v * ad[h * 64 + c];
    }
    asrc[idx] = s;
    adst[idx] = d;
}

__global__ void seg_init_k(float* agg, unsigned* mmax, float* den, int ND, int NH) {
    int i = blockIdx.x * blockDim.x + threadIdx.x;
    if (i < ND) agg[i] = 0.f;
    if (i < NH) { mmax[i] = fenc(-3.4e38f); den[i] = 0.f; }
}

__global__ void edge_score_k(const int* __restrict__ ei, const float* __restrict__ asrc,
                             const float* __restrict__ adst, float* __restrict__ e,
                             unsigned* __restrict__ mmax, int E, int N, int heads) {
    int idx = blockIdx.x * blockDim.x + threadIdx.x;
    if (idx >= (E + N) * heads) return;
    int t = idx / heads, h = idx % heads;
    int s = (t < E) ? ei[t] : (t - E);   // self loops appended after E edges
    int d = (t < E) ? ei[E + t] : (t - E);
    float v = lrelu(asrc[s * heads + h] + adst[d * heads + h]);
    e[idx] = v;
    atomicMax(&mmax[d * heads + h], fenc(v));
}

__global__ void edge_exp_k(const int* __restrict__ ei, float* __restrict__ e,
                           const unsigned* __restrict__ mmax, float* __restrict__ den,
                           int E, int N, int heads) {
    int idx = blockIdx.x * blockDim.x + threadIdx.x;
    if (idx >= (E + N) * heads) return;
    int t = idx / heads, h = idx % heads;
    int d = (t < E) ? ei[E + t] : (t - E);
    float ex = __expf(e[idx] - fdec(mmax[d * heads + h]));
    e[idx] = ex;
    atomicAdd(&den[d * heads + h], ex);
}

__global__ void edge_agg_k(const int* __restrict__ ei, const float* __restrict__ e,
                           const float* __restrict__ den, const float* __restrict__ hw,
                           float* __restrict__ agg, int E, int N, int heads) {
    int idx = blockIdx.x * blockDim.x + threadIdx.x;
    if (idx >= (E + N) * heads) return;
    int t = idx / heads, h = idx % heads;
    int s = (t < E) ? ei[t] : (t - E);
    int d = (t < E) ? ei[E + t] : (t - E);
    float coef = e[idx] / (den[d * heads + h] + 1e-16f);
    const float* srow = hw + (size_t)s * heads * 64 + h * 64;
    float* drow = agg + (size_t)d * heads * 64 + h * 64;
#pragma unroll 4
    for (int c = 0; c < 64; ++c) atomicAdd(&drow[c], srow[c] * coef);
}

// ---------------- BatchNorm batch statistics ----------------
__global__ void col_stats_k(const float* __restrict__ X, float* __restrict__ mu,
                            float* __restrict__ var, int N, int D) {
    __shared__ float sh[256], sh2[256];
    int d = blockIdx.x;
    float s = 0.f, s2 = 0.f;
    for (int n = threadIdx.x; n < N; n += blockDim.x) {
        float v = X[(size_t)n * D + d];
        s += v; s2 += v * v;
    }
    sh[threadIdx.x] = s; sh2[threadIdx.x] = s2;
    __syncthreads();
    for (int st = 128; st > 0; st >>= 1) {
        if ((int)threadIdx.x < st) {
            sh[threadIdx.x] += sh[threadIdx.x + st];
            sh2[threadIdx.x] += sh2[threadIdx.x + st];
        }
        __syncthreads();
    }
    if (threadIdx.x == 0) {
        float m = sh[0] / N;
        mu[d] = m;
        var[d] = sh2[0] / N - m * m;
    }
}

// out = leaky( BN(agg)*g + be + res )  (GAT bias cancels exactly in BN's mean subtraction)
__global__ void bn_res_act_k(const float* __restrict__ agg, const float* __restrict__ mu,
                             const float* __restrict__ var, const float* __restrict__ g,
                             const float* __restrict__ be, const float* __restrict__ res,
                             float* __restrict__ out, int N, int D) {
    int idx = blockIdx.x * blockDim.x + threadIdx.x;
    if (idx >= N * D) return;
    int d = idx % D;
    float v = (agg[idx] - mu[d]) * rsqrtf(var[d] + 1e-5f) * g[d] + be[d] + res[idx];
    out[idx] = lrelu(v);
}

__global__ void max3_k(const float* a, const float* b, const float* c, float* o, int n) {
    int i = blockIdx.x * blockDim.x + threadIdx.x;
    if (i < n) o[i] = fmaxf(fmaxf(a[i], b[i]), c[i]);
}

// Zero-pad weight [K][N] -> [K][Np]
__global__ void pad_w_k(const float* __restrict__ w, float* __restrict__ wp,
                        int K, int N, int Np) {
    int i = blockIdx.x * blockDim.x + threadIdx.x;
    if (i >= K * Np) return;
    int k = i / Np, n = i % Np;
    wp[i] = (n < N) ? w[(size_t)k * N + n] : 0.f;
}
// Slice padded [M][Np] -> [M][N] and add bias
__global__ void slice_bias_k(const float* __restrict__ src, const float* __restrict__ bias,
                             float* __restrict__ dst, int M, int N, int Np) {
    int i = blockIdx.x * blockDim.x + threadIdx.x;
    if (i >= M * N) return;
    int m = i / N, n = i % N;
    dst[i] = src[(size_t)m * Np + n] + bias[n];
}

// ---------------- Flash attention: 1 wave per 16 query rows per head (d=32) ----------------
// 64 keys per iteration: one cross-lane softmax reduction amortized over 8 WMMAs.
__global__ __launch_bounds__(32) void flash_attn_k(const float* __restrict__ Q,
                                                   const float* __restrict__ Km,
                                                   const float* __restrict__ Vm,
                                                   float* __restrict__ O, int N) {
    __shared__ _Float16 P[16 * 64];   // probability tile, transposed via LDS into A layout
    const int h = blockIdx.y;
    const int q0 = blockIdx.x * 16;
    const int lane = threadIdx.x & 31, r = lane & 15, hi = lane >> 4;
    const float scale = 0.1767766952966369f;   // 1/sqrt(32), folded into Q at load
    v16h aq = load_a_f4(Q, 64, q0, h * 32, scale);

    float mrow[8], lsum[8];
#pragma unroll
    for (int i = 0; i < 8; ++i) { mrow[i] = -3.4e38f; lsum[i] = 0.f; }
    v8f o0 = {}, o1 = {};

    for (int j0 = 0; j0 < N; j0 += 64) {
        // prefetch next iteration's K/V rows
        __builtin_prefetch(Km + (size_t)(j0 + 64 + r) * 64 + h * 32, 0, 3);
        __builtin_prefetch(Vm + (size_t)(j0 + 64 + r) * 64 + h * 32, 0, 3);

        // S = Q(16x32) @ K^T for 4 key tiles of 16.
        v8f s[4];
#pragma unroll
        for (int t = 0; t < 4; ++t) {
            v16h bk = load_b_contig(Km + (size_t)(j0 + 16 * t + r) * 64 + h * 32 + 16 * hi);
            v8f zz = {};
            s[t] = __builtin_amdgcn_wmma_f32_16x16x32_f16(false, aq, false, bk, (short)0, zz,
                                                          false, false);
        }

        // Online softmax per query row (row <-> (hi,i)); one DPP lane-reduction per row.
#pragma unroll
        for (int i = 0; i < 8; ++i) {
            float mx = fmaxf(fmaxf(s[0][i], s[1][i]), fmaxf(s[2][i], s[3][i]));
            mx = row16_max(mx);
            float mnew = fmaxf(mrow[i], mx);
            float corr = __expf(mrow[i] - mnew);
            float p0 = __expf(s[0][i] - mnew);
            float p1 = __expf(s[1][i] - mnew);
            float p2 = __expf(s[2][i] - mnew);
            float p3 = __expf(s[3][i] - mnew);
            float ps = row16_sum((p0 + p1) + (p2 + p3));
            lsum[i] = lsum[i] * corr + ps;
            mrow[i] = mnew;
            o0[i] *= corr; o1[i] *= corr;
            _Float16* prow = P + (i + 8 * hi) * 64;
            prow[r]      = (_Float16)p0;
            prow[16 + r] = (_Float16)p1;
            prow[32 + r] = (_Float16)p2;
            prow[48 + r] = (_Float16)p3;
        }
        __syncthreads();

        // O += P(16x64) @ V(64x32): two A operands from LDS, four V B-operands.
        v16h ap0, ap1;
        {
            const _Float16* pr = P + r * 64 + 16 * hi;
            v8h x0 = *(const v8h*)(pr + 0);    // keys 16hi+0..7
            v8h x1 = *(const v8h*)(pr + 8);    // keys 16hi+8..15
            v8h x2 = *(const v8h*)(pr + 32);   // keys 32+16hi+0..7
            v8h x3 = *(const v8h*)(pr + 40);   // keys 32+16hi+8..15
#pragma unroll
            for (int j = 0; j < 8; ++j) {
                ap0[j] = x0[j]; ap0[j + 8] = x1[j];
                ap1[j] = x2[j]; ap1[j + 8] = x3[j];
            }
        }
        v16h bv00, bv01, bv10, bv11;
#pragma unroll
        for (int j = 0; j < 16; ++j) {
            int kk = j + 16 * hi;   // key index within tile of 32
            const float* vb  = Vm + (size_t)(j0 + kk) * 64 + h * 32;
            const float* vb2 = vb + (size_t)32 * 64;
            bv00[j] = (_Float16)vb[r];
            bv01[j] = (_Float16)vb[16 + r];
            bv10[j] = (_Float16)vb2[r];
            bv11[j] = (_Float16)vb2[16 + r];
        }
        o0 = __builtin_amdgcn_wmma_f32_16x16x32_f16(false, ap0, false, bv00, (short)0, o0, false, false);
        o1 = __builtin_amdgcn_wmma_f32_16x16x32_f16(false, ap0, false, bv01, (short)0, o1, false, false);
        o0 = __builtin_amdgcn_wmma_f32_16x16x32_f16(false, ap1, false, bv10, (short)0, o0, false, false);
        o1 = __builtin_amdgcn_wmma_f32_16x16x32_f16(false, ap1, false, bv11, (short)0, o1, false, false);
        __syncthreads();
    }
#pragma unroll
    for (int i = 0; i < 8; ++i) {
        float inv = 1.f / (lsum[i] + 1e-20f);
        O[(size_t)(q0 + i + 8 * hi) * 64 + h * 32 + r]      = o0[i] * inv;
        O[(size_t)(q0 + i + 8 * hi) * 64 + h * 32 + 16 + r] = o1[i] * inv;
    }
}

// --------------------------------- driver ---------------------------------
extern "C" void kernel_launch(void* const* d_in, const int* in_sizes, int n_in,
                              void* d_out, int out_size, void* d_ws, size_t ws_size,
                              hipStream_t stream) {
    const int Nn = in_sizes[0] / 64;   // 8192 nodes
    const int E  = in_sizes[1] / 2;    // 262144 edges

    const float* x  = (const float*)d_in[0];
    const int*   ei = (const int*)d_in[1];
    const float* W[3]  = {(const float*)d_in[2],  (const float*)d_in[6],  (const float*)d_in[10]};
    const float* AS[3] = {(const float*)d_in[3],  (const float*)d_in[7],  (const float*)d_in[11]};
    const float* AD[3] = {(const float*)d_in[4],  (const float*)d_in[8],  (const float*)d_in[12]};
    const float* R[3]  = {(const float*)d_in[5],  (const float*)d_in[9],  (const float*)d_in[13]};
    const float* G[3]  = {(const float*)d_in[23], (const float*)d_in[24], (const float*)d_in[25]};
    const float* BE[3] = {(const float*)d_in[26], (const float*)d_in[27], (const float*)d_in[28]};
    const int heads[3] = {4, 4, 1};
    const int din[3]   = {64, 256, 256};

    // workspace partition (floats)
    float* p = (float*)d_ws;
    auto take = [&](size_t n) { float* q = p; p += n; return q; };
    float* h1   = take((size_t)Nn * 256);
    float* h2   = take((size_t)Nn * 256);
    float* h3   = take((size_t)Nn * 64);
    float* hw   = take((size_t)Nn * 256);
    float* res  = take((size_t)Nn * 256);
    float* agg  = take((size_t)Nn * 256);
    float* asrc = take((size_t)Nn * 4);
    float* adst = take((size_t)Nn * 4);
    float* eex  = take((size_t)(E + Nn) * 4);
    unsigned* mmax = (unsigned*)take((size_t)Nn * 4);
    float* den  = take((size_t)Nn * 4);
    float* mu   = take(256);
    float* var  = take(256);
    float* z    = take((size_t)Nn * 64);
    float* t0   = take((size_t)Nn * 64);
    float* t1   = take((size_t)Nn * 64);
    float* q_   = take((size_t)Nn * 64);
    float* k_   = take((size_t)Nn * 64);
    float* v_   = take((size_t)Nn * 64);
    float* ao   = take((size_t)Nn * 64);
    float* zo   = take((size_t)Nn * 64);
    float* m1o  = take((size_t)Nn * 64);
    float* m2o  = take((size_t)Nn * 32);
    float* wpad = take(32 * 16);              // m3w padded 32x16
    float* opad = take((size_t)Nn * 16);      // padded final output

    auto gemm = [&](const float* A, const float* B, const float* bias, float* C,
                    int M, int K, int Nc, int act) {
        if (Nc % 64 == 0)
            gemm_wmma_k<4><<<dim3(M / 16, Nc / 64), 32, 0, stream>>>(A, B, bias, C, M, K, Nc, act);
        else if (Nc % 32 == 0)
            gemm_wmma_k<2><<<dim3(M / 16, Nc / 32), 32, 0, stream>>>(A, B, bias, C, M, K, Nc, act);
        else
            gemm_wmma_k<1><<<dim3(M / 16, Nc / 16), 32, 0, stream>>>(A, B, bias, C, M, K, Nc, act);
    };

    // ---- 3 GAT layers ----
    float* outs[3] = {h1, h2, h3};
    const float* h = x;
    for (int l = 0; l < 3; ++l) {
        int H = heads[l], D = H * 64, K = din[l];
        gemm(h, W[l], nullptr, hw,  Nn, K, D, 0);
        gemm(h, R[l], nullptr, res, Nn, K, D, 0);
        int nh = Nn * H, nd = Nn * D, tot = (E + Nn) * H;
        gat_alpha_k<<<(nh + 255) / 256, 256, 0, stream>>>(hw, AS[l], AD[l], asrc, adst, Nn, H);
        seg_init_k<<<(nd + 255) / 256, 256, 0, stream>>>(agg, mmax, den, nd, nh);
        edge_score_k<<<(tot + 255) / 256, 256, 0, stream>>>(ei, asrc, adst, eex, mmax, E, Nn, H);
        edge_exp_k<<<(tot + 255) / 256, 256, 0, stream>>>(ei, eex, mmax, den, E, Nn, H);
        edge_agg_k<<<(tot + 255) / 256, 256, 0, stream>>>(ei, eex, den, hw, agg, E, Nn, H);
        col_stats_k<<<D, 256, 0, stream>>>(agg, mu, var, Nn, D);
        bn_res_act_k<<<(nd + 255) / 256, 256, 0, stream>>>(agg, mu, var, G[l], BE[l], res,
                                                           outs[l], Nn, D);
        h = outs[l];
    }

    // ---- JumpingKnowledge max ----
    gemm(h1, (const float*)d_in[14], nullptr, t0, Nn, 256, 64, 0);
    gemm(h2, (const float*)d_in[15], nullptr, t1, Nn, 256, 64, 0);
    int n64 = Nn * 64;
    max3_k<<<(n64 + 255) / 256, 256, 0, stream>>>(t0, t1, h3, z, n64);

    // ---- self-attention (2 heads, d=32) ----
    gemm(z, (const float*)d_in[16], (const float*)d_in[32], q_, Nn, 64, 64, 0);
    gemm(z, (const float*)d_in[17], (const float*)d_in[33], k_, Nn, 64, 64, 0);
    gemm(z, (const float*)d_in[18], (const float*)d_in[34], v_, Nn, 64, 64, 0);
    flash_attn_k<<<dim3(Nn / 16, 2), 32, 0, stream>>>(q_, k_, v_, ao, Nn);
    gemm(ao, (const float*)d_in[19], (const float*)d_in[35], zo, Nn, 64, 64, 0);

    // ---- MLP head ----
    gemm(zo,  (const float*)d_in[20], (const float*)d_in[36], m1o, Nn, 64, 64, 1);
    gemm(m1o, (const float*)d_in[21], (const float*)d_in[37], m2o, Nn, 64, 32, 1);
    // final 32->10: pad weight to 32x16, GEMM, then slice + bias
    pad_w_k<<<(32 * 16 + 255) / 256, 256, 0, stream>>>((const float*)d_in[22], wpad, 32, 10, 16);
    gemm(m2o, wpad, nullptr, opad, Nn, 32, 16, 0);
    slice_bias_k<<<(Nn * 10 + 255) / 256, 256, 0, stream>>>(opad, (const float*)d_in[38],
                                                            (float*)d_out, Nn, 10, 16);
}
